// MultiHeadAttention_21784074125490
// MI455X (gfx1250) — matrix operand order repaired and checked by reading, compiled, not run on hardware
//
#include <hip/hip_runtime.h>
#include <math.h>

// ---------------------------------------------------------------------------
// Types
// ---------------------------------------------------------------------------
typedef _Float16 half_t;
typedef __attribute__((ext_vector_type(16))) _Float16 v16h;
typedef __attribute__((ext_vector_type(8)))  _Float16 v8h;
typedef __attribute__((ext_vector_type(4)))  _Float16 v4h;
typedef __attribute__((ext_vector_type(8)))  float    v8f;
typedef __attribute__((ext_vector_type(4)))  float    v4f;
typedef __attribute__((ext_vector_type(4)))  unsigned int u32x4;
typedef __attribute__((ext_vector_type(8)))  int          i32x8;
typedef __attribute__((ext_vector_type(4)))  int          i32x4;

#define D_MODEL 1024
#define SEQ     2048
#define HEADS   16
#define HDIM    64
#define BATCH   4
#define M_TOT   (BATCH * SEQ)   // 8192
#define KVSTRIDE 72             // padded LDS stride (halves) for 64-wide tiles

// ---------------------------------------------------------------------------
// Helpers
// ---------------------------------------------------------------------------
__device__ __forceinline__ v8f wmma_f16(v16h a, v16h b, v8f c) {
    return __builtin_amdgcn_wmma_f32_16x16x32_f16(
        /*neg_a=*/false, a, /*neg_b=*/false, b,
        /*c_mod=*/(short)0, c, /*reuse_a=*/false, /*reuse_b=*/false);
}

// Build a 16-half fragment from two contiguous 8-half (16B) chunks.
__device__ __forceinline__ v16h ld_frag(const half_t* p0, const half_t* p1) {
    v8h a = *(const v8h*)p0;
    v8h b = *(const v8h*)p1;
    v16h r;
#pragma unroll
    for (int i = 0; i < 8; ++i) { r[i] = a[i]; r[i + 8] = b[i]; }
    return r;
}

__device__ __forceinline__ v4h load4_h(const float* p) {
    v4f f = *(const v4f*)p;
    v4h h;
#pragma unroll
    for (int i = 0; i < 4; ++i) h[i] = (_Float16)f[i];
    return h;
}
__device__ __forceinline__ v4h load4_h(const half_t* p) {
    return *(const v4h*)p;
}

// ---------------------------------------------------------------------------
// TDM: async DMA of a 64x64 f16 tile (row stride HDIM halves in global) into
// LDS with a 16-byte pad every 128 bytes -> effective LDS stride 72 halves.
// D# layout per CDNA5 ISA ch.8 (group0 128b, group1 256b, groups 2/3 zero).
// This toolchain exposes the 6-argument builtin form.
// ---------------------------------------------------------------------------
__device__ __forceinline__ void tdm_load_tile64(const half_t* gsrc,
                                                unsigned lds_byte_addr) {
    unsigned long long ga = (unsigned long long)(size_t)gsrc;
    u32x4 g0;
    g0[0] = 1u;                                   // count=1, user mode
    g0[1] = lds_byte_addr;                        // lds_addr
    g0[2] = (unsigned)ga;                         // global_addr[31:0]
    g0[3] = (unsigned)((ga >> 32) & 0x01FFFFFFu)  // global_addr[56:32]
          | (2u << 30);                           // type = 2 (image)
    i32x8 g1;
    g1[0] = (int)((1u << 16)     // data_size = 2 bytes
                | (1u << 20)     // pad_enable
                | (4u << 22)     // pad_interval: 2^(4+1)=32 DWORDs (128B)
                | (3u << 25));   // pad_amount: 3+1 = 4 DWORDs (16B)
    g1[1] = (int)((unsigned)HDIM << 16);          // tensor_dim0 = 64
    g1[2] = (int)((unsigned)SEQ << 16);           // tensor_dim1 = 2048
    g1[3] = (int)(64u << 16);                     // tile_dim0 = 64
    g1[4] = 64;                                   // tile_dim1 = 64
    g1[5] = HDIM;                                 // tensor_dim0_stride = 64
    g1[6] = 0;
    g1[7] = 0;
    i32x4 z4 = {0, 0, 0, 0};
    i32x8 z8 = {0, 0, 0, 0, 0, 0, 0, 0};
    __builtin_amdgcn_tensor_load_to_lds(g0, g1, z4, z4, z8, /*cpol=*/0);
}

// ---------------------------------------------------------------------------
// GEMM:  C[M x N] = A[M x K](f32 or f16) * W[K x N](f32) + bias
// Block tile: 128(M) x 128(N); 8 waves (4x2), each wave 32x64 via 2x4 WMMAs.
// Output: either f16 head-major [B,H,S,64] (outH) or f32 row-major (outF).
// ---------------------------------------------------------------------------
template <typename TA>
__global__ __launch_bounds__(256)
void gemm_proj_kernel(const TA* __restrict__ A, const float* __restrict__ W,
                      const float* __restrict__ bias,
                      half_t* __restrict__ outH, float* __restrict__ outF,
                      int Kdim, int Ndim) {
    __shared__ half_t As[128 * 40];  // 128 rows x K-step 32, padded stride 40
    __shared__ half_t Bs[128 * 40];  // 128 cols (N) x 32 (K), transposed

    const int tid    = threadIdx.x;
    const int lane   = tid & 31;
    const int wid    = tid >> 5;
    const int hi     = lane >> 4;
    const int ln15   = lane & 15;
    const int wave_m = wid >> 1;   // 0..3 -> 32 rows each
    const int wave_n = wid & 1;    // 0..1 -> 64 cols each

    const int row0 = blockIdx.y * 128;
    const int n0   = blockIdx.x * 128;

    v8f acc[2][4] = {};

    for (int k0 = 0; k0 < Kdim; k0 += 32) {
        // ---- stage A tile: 128x32 -> f16 LDS ----
#pragma unroll
        for (int it = 0; it < 4; ++it) {
            int e = tid * 4 + it * 1024;
            int r = e >> 5;
            int c = e & 31;
            v4h h = load4_h(A + (size_t)(row0 + r) * Kdim + k0 + c);
            *(v4h*)&As[r * 40 + c] = h;
        }
        // ---- stage W tile: 32(K) x 128(N) f32 -> transposed f16 LDS ----
#pragma unroll
        for (int it = 0; it < 4; ++it) {
            int e  = tid * 4 + it * 1024;
            int kk = e >> 7;
            int nn = e & 127;
            v4f f = *(const v4f*)(W + (size_t)(k0 + kk) * Ndim + n0 + nn);
#pragma unroll
            for (int q = 0; q < 4; ++q)
                Bs[(nn + q) * 40 + kk] = (_Float16)f[q];
        }
        if (k0 + 32 < Kdim)
            __builtin_prefetch((const void*)(A + (size_t)(row0 + (tid >> 1)) * Kdim + k0 + 32), 0, 1);

        __syncthreads();

        // ---- fragments + WMMA ----
        v16h af[2], bf[4];
#pragma unroll
        for (int mi = 0; mi < 2; ++mi) {
            const half_t* p = &As[(wave_m * 32 + mi * 16 + ln15) * 40];
            // A frag: k = j + (j&8) + 8*hi -> chunks at hi*8 and 16+hi*8
            af[mi] = ld_frag(p + hi * 8, p + 16 + hi * 8);
        }
#pragma unroll
        for (int ni = 0; ni < 4; ++ni) {
            const half_t* p = &Bs[(wave_n * 64 + ni * 16 + ln15) * 40];
            // B frag: k = j + (j&8) + 8*hi -> chunks at hi*8 and 16+hi*8
            bf[ni] = ld_frag(p + hi * 8, p + 16 + hi * 8);
        }
#pragma unroll
        for (int mi = 0; mi < 2; ++mi)
#pragma unroll
            for (int ni = 0; ni < 4; ++ni)
                acc[mi][ni] = wmma_f16(af[mi], bf[ni], acc[mi][ni]);

        __syncthreads();
    }

    // ---- epilogue ----
#pragma unroll
    for (int mi = 0; mi < 2; ++mi) {
#pragma unroll
        for (int ni = 0; ni < 4; ++ni) {
            int n_g = n0 + wave_n * 64 + ni * 16 + ln15;
            float bn = bias[n_g];
#pragma unroll
            for (int r = 0; r < 8; ++r) {
                int m_g = row0 + wave_m * 32 + mi * 16 + r + 8 * hi;
                float val = acc[mi][ni][r] + bn;
                if (outF) {
                    outF[(size_t)m_g * Ndim + n_g] = val;
                } else {
                    int b = m_g >> 11;          // / SEQ
                    int s = m_g & (SEQ - 1);
                    int h = n_g >> 6;           // / HDIM
                    int d = n_g & (HDIM - 1);
                    outH[(((size_t)b * HEADS + h) * SEQ + s) * HDIM + d] = (_Float16)val;
                }
            }
        }
    }
}

// ---------------------------------------------------------------------------
// Flash attention: Q,K,V f16 in [B*H, S, 64]; ctx f16 out in [B, S, D].
// 128 query rows per block, 8 waves x 16 rows, key blocks of 64.
// K tiles are double-buffered via the Tensor Data Mover (async DMA to LDS,
// pad feature produces the bank-conflict-free 72-half stride); V tiles are
// double-buffered with a manual transpose (TDM cannot transpose).
// ---------------------------------------------------------------------------
__global__ __launch_bounds__(256)
void attn_kernel(const half_t* __restrict__ Q, const half_t* __restrict__ K,
                 const half_t* __restrict__ V, half_t* __restrict__ ctx) {
    __shared__ half_t Ks[2][64 * KVSTRIDE];   // [key][d], padded (TDM-filled)
    __shared__ half_t Vt[2][64 * KVSTRIDE];   // [d][key], padded (transposed)
    __shared__ half_t Ps[8 * 16 * KVSTRIDE];  // per-wave P tiles

    const int tid  = threadIdx.x;
    const int lane = tid & 31;
    const int wid  = tid >> 5;
    const int hi   = lane >> 4;
    const int ln15 = lane & 15;

    const int bh = blockIdx.y;             // b*HEADS + h
    const int q0 = blockIdx.x * 128 + wid * 16;

    const half_t* Khead = K + (size_t)bh * SEQ * HDIM;

    // preload Q fragments (K-dim = hd = 64 -> 2 fragments of 32)
    const half_t* qrow = Q + ((size_t)bh * SEQ + q0 + ln15) * HDIM;
    v16h qf[2];
    qf[0] = ld_frag(qrow + hi * 8,      qrow + 16 + hi * 8);
    qf[1] = ld_frag(qrow + 32 + hi * 8, qrow + 48 + hi * 8);

    v8f acc[4] = {};
    float mrun[8], lrun[8];
#pragma unroll
    for (int r = 0; r < 8; ++r) { mrun[r] = -1.0e30f; lrun[r] = 0.0f; }

    const int pbase = wid * 16 * KVSTRIDE;

    // prime the pipeline: TDM load of first K tile into buffer 0
    if (wid == 0)
        tdm_load_tile64(Khead, (unsigned)(size_t)(const void*)&Ks[0][0]);

    for (int kb = 0; kb < SEQ; kb += 64) {
        const int buf = (kb >> 6) & 1;

        // ---- stage V tile (transposed) into current buffer ----
#pragma unroll
        for (int it = 0; it < 2; ++it) {
            int e  = tid * 8 + it * 2048;
            int kk = e >> 6;
            int d  = e & 63;
            v8h tv = *(const v8h*)(V + ((size_t)bh * SEQ + kb + kk) * HDIM + d);
#pragma unroll
            for (int q = 0; q < 8; ++q)
                Vt[buf][(d + q) * KVSTRIDE + kk] = tv[q];
        }

        // ---- wait for the TDM that filled Ks[buf], rendezvous ----
        if (wid == 0)
            __builtin_amdgcn_s_wait_tensorcnt(0);
        __syncthreads();

        // ---- kick off async DMA of the NEXT K tile while we compute ----
        if (wid == 0 && kb + 64 < SEQ)
            tdm_load_tile64(Khead + (size_t)(kb + 64) * HDIM,
                            (unsigned)(size_t)(const void*)&Ks[buf ^ 1][0]);

        // ---- scores: S[16 x 64] = Q * K^T, scaled ----
        v8f st[4];
#pragma unroll
        for (int t = 0; t < 4; ++t) {
            st[t] = (v8f){};
#pragma unroll
            for (int ds = 0; ds < 2; ++ds) {
                const half_t* p = &Ks[buf][(t * 16 + ln15) * KVSTRIDE + ds * 32];
                v16h kf = ld_frag(p + hi * 8, p + 16 + hi * 8);
                st[t] = wmma_f16(qf[ds], kf, st[t]);
            }
#pragma unroll
            for (int r = 0; r < 8; ++r) st[t][r] = st[t][r] * 0.125f; // 1/sqrt(64)
        }

        // ---- online softmax (row stats across 16-lane halves) ----
        float rm[8];
#pragma unroll
        for (int r = 0; r < 8; ++r) {
            float v = st[0][r];
            v = fmaxf(v, st[1][r]); v = fmaxf(v, st[2][r]); v = fmaxf(v, st[3][r]);
#pragma unroll
            for (int m = 1; m < 16; m <<= 1)
                v = fmaxf(v, __shfl_xor(v, m, 16));
            rm[r] = v;
        }
        float rs[8];
#pragma unroll
        for (int r = 0; r < 8; ++r) {
            float mnew = fmaxf(mrun[r], rm[r]);
            float scl  = __expf(mrun[r] - mnew);
            mrun[r] = mnew;
            lrun[r] *= scl;
#pragma unroll
            for (int t = 0; t < 4; ++t) acc[t][r] = acc[t][r] * scl;
            rs[r] = 0.0f;
        }
        // P = exp(S - mnew) -> wave-private LDS region in [row][key] order
#pragma unroll
        for (int t = 0; t < 4; ++t) {
#pragma unroll
            for (int r = 0; r < 8; ++r) {
                float p = __expf(st[t][r] - mrun[r]);
                rs[r] += p;
                Ps[pbase + (r + 8 * hi) * KVSTRIDE + t * 16 + ln15] = (_Float16)p;
            }
        }
#pragma unroll
        for (int r = 0; r < 8; ++r) {
            float v = rs[r];
#pragma unroll
            for (int m = 1; m < 16; m <<= 1)
                v += __shfl_xor(v, m, 16);
            lrun[r] += v;
        }

        // ---- ctx += P * V ----
#pragma unroll
        for (int ks = 0; ks < 2; ++ks) {
            const half_t* pp = &Ps[pbase + ln15 * KVSTRIDE + ks * 32 + hi * 8];
            v16h pf = ld_frag(pp, pp + 16);
#pragma unroll
            for (int t = 0; t < 4; ++t) {
                const half_t* vp = &Vt[buf][(t * 16 + ln15) * KVSTRIDE + ks * 32];
                v16h vf = ld_frag(vp + hi * 8, vp + 16 + hi * 8);
                acc[t] = wmma_f16(pf, vf, acc[t]);
            }
        }
        // no trailing barrier: double buffering + the rendezvous barrier above
        // guarantee no wave touches a buffer another wave still needs.
    }

    // ---- finalize: divide by l, write ctx in [B, S, D] (head-contig cols) ----
    const int b = bh >> 4;
    const int h = bh & 15;
#pragma unroll
    for (int t = 0; t < 4; ++t) {
#pragma unroll
        for (int r = 0; r < 8; ++r) {
            int qg = q0 + r + 8 * hi;
            int d  = t * 16 + ln15;
            float val = acc[t][r] / lrun[r];
            ctx[((size_t)b * SEQ + qg) * D_MODEL + h * HDIM + d] = (_Float16)val;
        }
    }
}

// ---------------------------------------------------------------------------
// Launch
// ---------------------------------------------------------------------------
extern "C" void kernel_launch(void* const* d_in, const int* in_sizes, int n_in,
                              void* d_out, int out_size, void* d_ws, size_t ws_size,
                              hipStream_t stream) {
    (void)in_sizes; (void)n_in; (void)out_size; (void)ws_size;

    const float* query = (const float*)d_in[0];
    const float* key   = (const float*)d_in[1];
    const float* value = (const float*)d_in[2];
    const float* Wq = (const float*)d_in[3];
    const float* bq = (const float*)d_in[4];
    const float* Wk = (const float*)d_in[5];
    const float* bk = (const float*)d_in[6];
    const float* Wv = (const float*)d_in[7];
    const float* bv = (const float*)d_in[8];
    const float* Wo = (const float*)d_in[9];
    const float* bo = (const float*)d_in[10];
    float* out = (float*)d_out;

    const size_t ELEMS = (size_t)M_TOT * D_MODEL;  // 8M halves each
    half_t* q_hf  = (half_t*)d_ws;
    half_t* k_hf  = q_hf + ELEMS;
    half_t* v_hf  = k_hf + ELEMS;
    half_t* ctx_h = v_hf + ELEMS;

    dim3 gG(D_MODEL / 128, M_TOT / 128);  // 8 x 64
    dim3 gA(SEQ / 128, BATCH * HEADS);    // 16 x 64

    gemm_proj_kernel<float><<<gG, 256, 0, stream>>>(query, Wq, bq, q_hf, nullptr, D_MODEL, D_MODEL);
    gemm_proj_kernel<float><<<gG, 256, 0, stream>>>(key,   Wk, bk, k_hf, nullptr, D_MODEL, D_MODEL);
    gemm_proj_kernel<float><<<gG, 256, 0, stream>>>(value, Wv, bv, v_hf, nullptr, D_MODEL, D_MODEL);

    attn_kernel<<<gA, 256, 0, stream>>>(q_hf, k_hf, v_hf, ctx_h);

    gemm_proj_kernel<_Float16><<<gG, 256, 0, stream>>>(ctx_h, Wo, bo, nullptr, out, D_MODEL, D_MODEL);
}
